// HRPA_979252544026
// MI455X (gfx1250) — matrix operand order repaired and checked by reading, not compiled, not run on hardware
//
#include <hip/hip_runtime.h>
#include <hip/hip_bf16.h>

// Problem constants from the reference:  B=256, N=1024, M=128, C=512
#define B_  256
#define N_  1024
#define M_  128
#define C_  512
#define NT  128        // N-tile per block
#define KC  64         // K-chunk staged in LDS per iteration
#define LDW 72         // padded LDS row length in halves (144 B, 16B-aligned stride)

typedef __attribute__((ext_vector_type(16))) _Float16 v16h;
typedef __attribute__((ext_vector_type(8)))  _Float16 v8h;
typedef __attribute__((ext_vector_type(4)))  _Float16 v4h;
typedef __attribute__((ext_vector_type(8)))  float    v8f;

union ABu { v16h v; v8h h[2]; };

// ---------------------------------------------------------------------------
// Kernel 1: per (batch, n-tile) block computes a 128x128 sim tile via
// v_wmma_f32_16x16x32_f16, then partial row-max (to ws) and the full masked
// column contribution for this tile (to ws).
// ---------------------------------------------------------------------------
__global__ __launch_bounds__(256)
void HRPA_sim_tile_kernel(const float* __restrict__ patch,
                          const float* __restrict__ word,
                          const float* __restrict__ mask,
                          float* __restrict__ rowPartial,      // [B][8][128]
                          float* __restrict__ colsumPartial)   // [B][8]
{
    __shared__ _Float16 sWord [M_ * LDW];   // 128 x 64 f16 (padded)  18 KB
    __shared__ _Float16 sPatch[NT * LDW];   // 128 x 64 f16 (padded)  18 KB
    __shared__ float    colpart[8 * NT];    // per-wave column maxes   4 KB
    __shared__ float    redbuf[8];

    const int nt   = blockIdx.x;            // n-tile index (0..7)
    const int b    = blockIdx.y;            // batch
    const int tid  = threadIdx.x;
    const int lane = tid & 31;
    const int wv   = tid >> 5;              // wave id (0..7) -> M rows [16wv,16wv+16)
    const int n0   = nt * NT;

    const float* wordB  = word  + (size_t)b * M_ * C_;
    const float* patchB = patch + (size_t)b * N_ * C_ + (size_t)n0 * C_;

    // per-lane penalty for the row path: depends only on n = 16j + (lane&15)
    float pen[8];
#pragma unroll
    for (int j = 0; j < 8; ++j) {
        float mv = mask[(size_t)b * N_ + n0 + j * 16 + (lane & 15)];
        pen[j] = 1000.0f * (1.0f - mv);
    }

    v8f acc[8];
#pragma unroll
    for (int j = 0; j < 8; ++j)
#pragma unroll
        for (int i = 0; i < 8; ++i) acc[j][i] = 0.0f;

    for (int kc = 0; kc < C_; kc += KC) {
        __syncthreads();   // LDS reuse from previous compute phase

        // ---- stage 128x64 f32 of word & patch into LDS as f16 (coalesced) --
#pragma unroll
        for (int it = 0; it < 8; ++it) {
            int idx4 = tid + it * 256;            // 0..2047 float4 slots
            int row  = idx4 >> 4;                 // 16 float4 per 64-wide row
            int c4   = (idx4 & 15) * 4;

            float4 fw = *(const float4*)(wordB + (size_t)row * C_ + kc + c4);
            v4h hw = { (_Float16)fw.x, (_Float16)fw.y, (_Float16)fw.z, (_Float16)fw.w };
            *(v4h*)&sWord[row * LDW + c4] = hw;

            float4 fp = *(const float4*)(patchB + (size_t)row * C_ + kc + c4);
            v4h hp = { (_Float16)fp.x, (_Float16)fp.y, (_Float16)fp.z, (_Float16)fp.w };
            *(v4h*)&sPatch[row * LDW + c4] = hp;

            // prefetch next K-chunk (global_prefetch_b8)
            if (kc + KC < C_) {
                __builtin_prefetch(wordB  + (size_t)row * C_ + kc + KC + c4, 0, 1);
                __builtin_prefetch(patchB + (size_t)row * C_ + kc + KC + c4, 0, 1);
            }
        }
        __syncthreads();

        // ---- compute: 2 K-steps of 32, 8 N-subtiles each ------------------
#pragma unroll
        for (int ks = 0; ks < 2; ++ks) {
            const int kbase = ks * 32;

            // A fragment (16x32 f16): lane = M row; halves per ISA layout
            const int am = (wv << 4) + (lane & 15);
            const int ak = kbase + ((lane & 16) ? 8 : 0);
            ABu a;
            a.h[0] = *(const v8h*)&sWord[am * LDW + ak];        // K = ak..ak+7
            a.h[1] = *(const v8h*)&sWord[am * LDW + ak + 16];   // K = ak+16..ak+23

#pragma unroll
            for (int j = 0; j < 8; ++j) {
                // B fragment (32x16 f16): lane = N col; halves per ISA layout (same K order as A)
                const int bn = j * 16 + (lane & 15);
                const int bk = kbase + ((lane & 16) ? 8 : 0);
                ABu bb;
                bb.h[0] = *(const v8h*)&sPatch[bn * LDW + bk];
                bb.h[1] = *(const v8h*)&sPatch[bn * LDW + bk + 16];

                acc[j] = __builtin_amdgcn_wmma_f32_16x16x32_f16(
                    /*neg_a=*/false, a.v, /*neg_b=*/false, bb.v,
                    /*c_mod=*/(short)0, acc[j],
                    /*reuse_a=*/false, /*reuse_b=*/false);
            }
        }
    }

    // ---- epilogue: leaky relu, row max (partial), column max (full) -------
    float rmax[8];
#pragma unroll
    for (int i = 0; i < 8; ++i) rmax[i] = -3.4e38f;

#pragma unroll
    for (int j = 0; j < 8; ++j) {
        float cm = -3.4e38f;
#pragma unroll
        for (int i = 0; i < 8; ++i) {
            float s = acc[j][i];
            s = (s >= 0.0f) ? s : 0.1f * s;        // leaky_relu(0.1)
            cm = fmaxf(cm, s);                     // column path (no penalty)
            rmax[i] = fmaxf(rmax[i], s - pen[j]);  // row path (masked penalty)
        }
        // lanes L and L+16 hold the same n (different M halves): combine
        cm = fmaxf(cm, __shfl_xor(cm, 16));
        if (lane < 16) colpart[wv * NT + j * 16 + lane] = cm;
    }

    // row max across the 16 n's held by a 16-lane group
#pragma unroll
    for (int i = 0; i < 8; ++i) {
        float r = rmax[i];
        r = fmaxf(r, __shfl_xor(r, 1));
        r = fmaxf(r, __shfl_xor(r, 2));
        r = fmaxf(r, __shfl_xor(r, 4));
        r = fmaxf(r, __shfl_xor(r, 8));
        rmax[i] = r;
    }
    if ((lane & 15) == 0) {
        // lane 0 -> M = 16wv+0..7 ; lane 16 -> M = 16wv+8..15
        const int mbase = (wv << 4) + ((lane >> 4) << 3);
        float* rp = rowPartial + (size_t)(b * 8 + nt) * M_ + mbase;
#pragma unroll
        for (int i = 0; i < 8; ++i) rp[i] = rmax[i];
    }
    __syncthreads();

    // column: max over the 8 waves, then masked sum over this tile's 128 n's
    float cv = 0.0f;
    if (tid < NT) {
        float c = colpart[tid];
#pragma unroll
        for (int w = 1; w < 8; ++w) c = fmaxf(c, colpart[w * NT + tid]);
        cv = c * mask[(size_t)b * N_ + n0 + tid];
    }
    for (int off = 16; off; off >>= 1) cv += __shfl_xor(cv, off);
    if (lane == 0) redbuf[wv] = cv;
    __syncthreads();
    if (tid == 0) {
        float s = 0.0f;
#pragma unroll
        for (int w = 0; w < 8; ++w) s += redbuf[w];
        colsumPartial[b * 8 + nt] = s;
    }
}

// ---------------------------------------------------------------------------
// Kernel 2: per-batch final reduction.
// ---------------------------------------------------------------------------
__global__ __launch_bounds__(128)
void HRPA_reduce_kernel(const float* __restrict__ rowPartial,
                        const float* __restrict__ colsumPartial,
                        const float* __restrict__ mask,
                        float* __restrict__ out)
{
    const int b = blockIdx.x;
    const int t = threadIdx.x;            // 128 threads = one per word m

    float r = -3.4e38f;
#pragma unroll
    for (int ntile = 0; ntile < 8; ++ntile)
        r = fmaxf(r, rowPartial[(size_t)(b * 8 + ntile) * M_ + t]);

    float ms = 0.0f;
#pragma unroll
    for (int i = 0; i < 8; ++i)
        ms += mask[(size_t)b * N_ + t + 128 * i];

    __shared__ float red[2][4];
    float rs = r, mss = ms;
    for (int off = 16; off; off >>= 1) {
        rs  += __shfl_xor(rs, off);
        mss += __shfl_xor(mss, off);
    }
    if ((t & 31) == 0) { red[0][t >> 5] = rs; red[1][t >> 5] = mss; }
    __syncthreads();
    if (t == 0) {
        float rowsum = 0.0f, msum = 0.0f;
#pragma unroll
        for (int w = 0; w < 4; ++w) { rowsum += red[0][w]; msum += red[1][w]; }
        float cs = 0.0f;
#pragma unroll
        for (int ntile = 0; ntile < 8; ++ntile) cs += colsumPartial[b * 8 + ntile];
        out[b] = rowsum * (1.0f / (float)M_) + cs / (msum + 1e-8f);
    }
}

// ---------------------------------------------------------------------------
extern "C" void kernel_launch(void* const* d_in, const int* in_sizes, int n_in,
                              void* d_out, int out_size, void* d_ws, size_t ws_size,
                              hipStream_t stream) {
    const float* patch = (const float*)d_in[0];   // (B, N, C) f32
    const float* word  = (const float*)d_in[1];   // (B, M, C) f32
    const float* mask  = (const float*)d_in[2];   // (B, N)    f32
    float* out = (float*)d_out;                   // (B, 1)    f32

    float* rowPartial    = (float*)d_ws;                                   // B*8*128 f32 = 1 MB
    float* colsumPartial = (float*)((char*)d_ws +
                           (size_t)B_ * 8 * M_ * sizeof(float));           // B*8 f32

    dim3 grid1(N_ / NT, B_);     // (8, 256)
    HRPA_sim_tile_kernel<<<grid1, 256, 0, stream>>>(patch, word, mask,
                                                    rowPartial, colsumPartial);
    HRPA_reduce_kernel<<<B_, 128, 0, stream>>>(rowPartial, colsumPartial, mask, out);
}
